// peepLSTM_15255723835573
// MI455X (gfx1250) — matrix operand order repaired and checked by reading, not compiled, not run on hardware
//
#include <hip/hip_runtime.h>
#include <hip/hip_bf16.h>

#define T_SEQ 256
#define EMB   512
#define HID   1024
#define BATCH 128
#define NCLS  10
#define NWG   32     // cooperative scan WGs; each owns 2x16 columns of H
#define WGSZ  256    // 8 waves (wave32); waves 0-3 -> N-tile 0, 4-7 -> N-tile 1

typedef __attribute__((ext_vector_type(16))) __bf16 v16bf;
typedef __attribute__((ext_vector_type(8)))  float  v8f;

union Frag16 { v16bf v; uint4 u[2]; };

__device__ __forceinline__ unsigned short f2bf(float f) {
  unsigned int x = __float_as_uint(f);
  x += 0x7FFFu + ((x >> 16) & 1u);          // round-to-nearest-even
  return (unsigned short)(x >> 16);
}
// fast sigmoid/tanh: v_exp_f32 + v_rcp_f32, no IEEE divide sequence
__device__ __forceinline__ float sigm_fast(float x) {
  return __builtin_amdgcn_rcpf(1.0f + __expf(-x));
}
__device__ __forceinline__ float tanh_fast(float x) {
  return fmaf(2.0f, sigm_fast(2.0f * x), -1.0f);
}

// -------- Phase 1a: gate lookup table  table[g][v][h] = (emb @ W_gx)[v,h] + b_g[h]
// g order: f,i,o,c ; for g==c we store sigmoid(value) directly (only ever used sigmoided).
__global__ void prep_table_k(const float* __restrict__ emb,
                             const float* __restrict__ Wfx, const float* __restrict__ Wix,
                             const float* __restrict__ Wox, const float* __restrict__ Wcx,
                             const float* __restrict__ bf,  const float* __restrict__ bi,
                             const float* __restrict__ bo,  const float* __restrict__ bc,
                             float* __restrict__ tbl) {
  int id = blockIdx.x * blockDim.x + threadIdx.x;
  if (id >= 4 * 3 * HID) return;
  int h = id % HID;
  int v = (id / HID) % 3;
  int g = id / (3 * HID);
  const float* W = (g == 0) ? Wfx : (g == 1) ? Wix : (g == 2) ? Wox : Wcx;
  const float* b = (g == 0) ? bf  : (g == 1) ? bi  : (g == 2) ? bo  : bc;
  float acc = b[h];
  for (int e = 0; e < EMB; ++e) acc = fmaf(emb[v * EMB + e], W[e * HID + h], acc);
  if (g == 3) acc = sigm_fast(acc);
  tbl[id] = acc;
}

// -------- Phase 1b: pack W_fh/W_ih/W_oh into bf16 WMMA B-fragment layout.
// pw[g][n_tile(64)][kk(32)][lane(32)][e(16)] ; B elem: N = n_tile*16 + lane%16,
// K = kk*32 + (lane<16 ? 0 : 16) + e  (lane data = 32 contiguous bytes).
__global__ void pack_w_k(const float* __restrict__ Wfh, const float* __restrict__ Wih,
                         const float* __restrict__ Woh, unsigned short* __restrict__ pw) {
  int id = blockIdx.x * blockDim.x + threadIdx.x;
  if (id >= 3 * 64 * 32 * 32 * 16) return;
  int e    = id & 15;
  int lane = (id >> 4) & 31;
  int kk   = (id >> 9) & 31;
  int nt   = (id >> 14) & 63;
  int g    = id >> 20;
  int k = kk * 32 + ((lane >> 4) << 4) + e;
  int n = nt * 16 + (lane & 15);
  const float* W = (g == 0) ? Wfh : (g == 1) ? Wih : Woh;
  pw[id] = f2bf(W[(size_t)k * HID + n]);
}

// -------- Phase 2: persistent cooperative peephole-LSTM scan.
// c kept in fp32 registers per owning lane; only the bf16 mirror is published per step.
__global__ void __launch_bounds__(WGSZ) lstm_scan_k(
    const int* __restrict__ x, const float* __restrict__ tbl,
    const unsigned short* __restrict__ pw,
    unsigned short* __restrict__ cb,  // bf16 cell state, 2 buffers of BATCH*HID (buf0 zeroed)
    float* __restrict__ hT,
    unsigned int* __restrict__ bar) { // T_SEQ grid-barrier counters (pre-zeroed)
  extern __shared__ char smem[];
  unsigned short* ldsW = (unsigned short*)smem;          // [2 nt][3 g][32 kk][32 lane][16] = 192 KB
  unsigned char*  ldsX = (unsigned char*)(smem + 196608); // x transposed: [t][b] bytes = 32 KB

  const int tid  = threadIdx.x;
  const int lane = tid & 31;
  const int wave = tid >> 5;
  const int bx   = blockIdx.x;

  // Stage this WG's two 16-column weight slices (f,i,o) into LDS once.
  {
    const uint4* src = (const uint4*)pw;
    uint4* dst = (uint4*)ldsW;
    for (int j = tid; j < 6 * 2048; j += WGSZ) {         // 2048 uint4 per (nt,g) slice
      int slice = j >> 11, jj = j & 2047;
      int ntl = slice / 3, g = slice % 3;
      dst[j] = src[(size_t)(g * 64 + ((bx << 1) + ntl)) * 2048 + jj];
    }
  }
  // Stage x transposed as bytes: ldsX[t*BATCH + b].
  if (tid < BATCH) {
    const int4* xr = (const int4*)(x + tid * T_SEQ);
    for (int tt = 0; tt < T_SEQ / 4; ++tt) {
      int4 v = xr[tt];
      ldsX[(tt * 4 + 0) * BATCH + tid] = (unsigned char)v.x;
      ldsX[(tt * 4 + 1) * BATCH + tid] = (unsigned char)v.y;
      ldsX[(tt * 4 + 2) * BATCH + tid] = (unsigned char)v.z;
      ldsX[(tt * 4 + 3) * BATCH + tid] = (unsigned char)v.w;
    }
  }
  __syncthreads();

  const int lane15 = lane & 15;
  const int half   = lane >> 4;
  const int ntl    = wave >> 2;                          // local N-tile (0/1)
  const int n      = (((bx << 1) + ntl) << 4) + lane15;  // output column (H index)
  const int mbase  = (wave & 3) << 5;                    // 32 batch rows per wave
  const int m0     = mbase + lane15;                     // A row, tile 0 (tile 1 = +16)
  const int rb0    = mbase + (half << 3);                // C/D rows: tile0 r -> rb0+r
  const int rb1    = rb0 + 16;

  // Preload all 12 possible gate-table values for this column (VOCAB=3).
  const float tF0 = tbl[0 * HID + n], tF1 = tbl[1 * HID + n],  tF2 = tbl[2 * HID + n];
  const float tI0 = tbl[3 * HID + n], tI1 = tbl[4 * HID + n],  tI2 = tbl[5 * HID + n];
  const float tO0 = tbl[6 * HID + n], tO1 = tbl[7 * HID + n],  tO2 = tbl[8 * HID + n];
  const float tC0 = tbl[9 * HID + n], tC1 = tbl[10 * HID + n], tC2 = tbl[11 * HID + n];

  float c0reg[8], c1reg[8];
  #pragma unroll
  for (int r = 0; r < 8; ++r) { c0reg[r] = 0.f; c1reg[r] = 0.f; }

  for (int t = 0; t < T_SEQ; ++t) {
    const int cur = t & 1, nxt = cur ^ 1;
    const unsigned short* cbc = cb + (size_t)cur * (BATCH * HID);
    unsigned short*       cbn = cb + (size_t)nxt * (BATCH * HID);

    v8f accF0 = {0.f,0.f,0.f,0.f,0.f,0.f,0.f,0.f};
    v8f accI0 = accF0, accO0 = accF0, accF1 = accF0, accI1 = accF0, accO1 = accF0;

    const unsigned short* a0row = cbc + (size_t)m0 * HID + (half << 4);
    const unsigned short* a1row = a0row + 16 * HID;
    #pragma unroll 2
    for (int kk = 0; kk < 32; ++kk) {
      Frag16 a0, a1, wf, wi, wo;
      const uint4* p0 = (const uint4*)(a0row + kk * 32);
      a0.u[0] = p0[0]; a0.u[1] = p0[1];
      const uint4* p1 = (const uint4*)(a1row + kk * 32);
      a1.u[0] = p1[0]; a1.u[1] = p1[1];
      const uint4* b0 = (const uint4*)(ldsW + (size_t)((ntl * 3 + 0) * 32 + kk) * 512 + lane * 16);
      const uint4* b1 = (const uint4*)(ldsW + (size_t)((ntl * 3 + 1) * 32 + kk) * 512 + lane * 16);
      const uint4* b2 = (const uint4*)(ldsW + (size_t)((ntl * 3 + 2) * 32 + kk) * 512 + lane * 16);
      wf.u[0] = b0[0]; wf.u[1] = b0[1];
      wi.u[0] = b1[0]; wi.u[1] = b1[1];
      wo.u[0] = b2[0]; wo.u[1] = b2[1];
      accF0 = __builtin_amdgcn_wmma_f32_16x16x32_bf16(false, a0.v, false, wf.v, (short)0, accF0, false, false);
      accI0 = __builtin_amdgcn_wmma_f32_16x16x32_bf16(false, a0.v, false, wi.v, (short)0, accI0, false, false);
      accO0 = __builtin_amdgcn_wmma_f32_16x16x32_bf16(false, a0.v, false, wo.v, (short)0, accO0, false, false);
      accF1 = __builtin_amdgcn_wmma_f32_16x16x32_bf16(false, a1.v, false, wf.v, (short)0, accF1, false, false);
      accI1 = __builtin_amdgcn_wmma_f32_16x16x32_bf16(false, a1.v, false, wi.v, (short)0, accI1, false, false);
      accO1 = __builtin_amdgcn_wmma_f32_16x16x32_bf16(false, a1.v, false, wo.v, (short)0, accO1, false, false);
    }

    // Elementwise update for the two 16x16 tiles (C/D layout: M = r (+8 for upper lanes)).
    const unsigned char* xp = ldsX + t * BATCH;
    uint2 xw0 = *(const uint2*)(xp + rb0);
    uint2 xw1 = *(const uint2*)(xp + rb1);

    #pragma unroll
    for (int tile = 0; tile < 2; ++tile) {
      v8f aF = tile ? accF1 : accF0;
      v8f aI = tile ? accI1 : accI0;
      v8f aO = tile ? accO1 : accO0;
      float* creg = tile ? c1reg : c0reg;
      uint2  xw   = tile ? xw1 : xw0;
      int    rb   = tile ? rb1 : rb0;
      #pragma unroll
      for (int r = 0; r < 8; ++r) {
        unsigned int xv = ((r < 4 ? (xw.x >> (8 * r)) : (xw.y >> (8 * (r - 4)))) & 0xFFu);
        float gF = (xv == 0u) ? tF0 : ((xv == 1u) ? tF1 : tF2);
        float gI = (xv == 0u) ? tI0 : ((xv == 1u) ? tI1 : tI2);
        float gO = (xv == 0u) ? tO0 : ((xv == 1u) ? tO1 : tO2);
        float gC = (xv == 0u) ? tC0 : ((xv == 1u) ? tC1 : tC2);
        float fg = sigm_fast(gF + aF[r]);
        float ig = sigm_fast(gI + aI[r]);
        float cn = fmaf(gC, ig, creg[r] * fg);           // sigmoid(gc) baked into table
        creg[r] = cn;
        cbn[(size_t)(rb + r) * HID + n] = f2bf(cn);
        if (t == T_SEQ - 1) {
          float og = sigm_fast(gO + aO[r]);
          hT[(size_t)(rb + r) * HID + n] = tanh_fast(cn) * og;
        } else {
          (void)gO; (void)aO;
        }
      }
    }

    // Grid-wide barrier: all WGs finish step t before anyone starts step t+1.
    __threadfence();
    __syncthreads();
    if (tid == 0) {
      atomicAdd(&bar[t], 1u);
      volatile unsigned int* p = bar + t;
      while (*p < (unsigned)NWG) __builtin_amdgcn_s_sleep(1);
    }
    __syncthreads();
    __threadfence();
  }
}

// -------- Phase 3: head  p = h_T @ W_ph + b_p ; log_softmax over 10 classes.
__global__ void head_k(const float* __restrict__ hT, const float* __restrict__ Wph,
                       const float* __restrict__ bp, float* __restrict__ out) {
  __shared__ float p[NCLS];
  __shared__ float red[2];
  int b = blockIdx.x, j = threadIdx.x;
  if (j < NCLS) {
    float acc = bp[j];
    for (int k = 0; k < HID; ++k) acc = fmaf(hT[(size_t)b * HID + k], Wph[k * NCLS + j], acc);
    p[j] = acc;
  }
  __syncthreads();
  if (j == 0) {
    float m = p[0];
    for (int q = 1; q < NCLS; ++q) m = fmaxf(m, p[q]);
    float s = 0.f;
    for (int q = 0; q < NCLS; ++q) s += __expf(p[q] - m);
    red[0] = m; red[1] = __logf(s);
  }
  __syncthreads();
  if (j < NCLS) out[b * NCLS + j] = p[j] - red[0] - red[1];
}

extern "C" void kernel_launch(void* const* d_in, const int* in_sizes, int n_in,
                              void* d_out, int out_size, void* d_ws, size_t ws_size,
                              hipStream_t stream) {
  const int*   x   = (const int*)d_in[0];
  const float* emb = (const float*)d_in[1];
  const float* Wfx = (const float*)d_in[2];
  const float* Wfh = (const float*)d_in[3];
  const float* bf  = (const float*)d_in[4];
  const float* Wix = (const float*)d_in[5];
  const float* Wih = (const float*)d_in[6];
  const float* bi  = (const float*)d_in[7];
  const float* Wox = (const float*)d_in[8];
  const float* Woh = (const float*)d_in[9];
  const float* bo  = (const float*)d_in[10];
  const float* Wcx = (const float*)d_in[11];
  const float* bc  = (const float*)d_in[12];
  const float* Wph = (const float*)d_in[13];
  const float* bp  = (const float*)d_in[14];
  float* out = (float*)d_out;

  // Workspace layout (bytes):
  //   tbl @ 0        : 48 KB   (4*3*1024 f32)
  //   pw  @ 49152    : 6 MB    (packed bf16 recurrent weights)
  //   cb  @ 6340608  : 512 KB  (2x bf16 cell-state buffers)
  //   bar @ 6864896  : 1 KB    (256 barrier counters)
  //   hT  @ 6865920  : 512 KB
  char* ws = (char*)d_ws;
  float*          tbl = (float*)ws;
  unsigned short* pw  = (unsigned short*)(ws + 49152);
  unsigned short* cb  = (unsigned short*)(ws + 6340608);
  unsigned int*   bar = (unsigned int*)(ws + 6864896);
  float*          hT  = (float*)(ws + 6865920);

  // Zero c0 (both bf16 ping-pong buffers) and the barrier counters each launch.
  hipMemsetAsync(cb, 0, 524288 + 1024, stream);

  prep_table_k<<<(4 * 3 * HID + 255) / 256, 256, 0, stream>>>(
      emb, Wfx, Wix, Wox, Wcx, bf, bi, bo, bc, tbl);
  pack_w_k<<<(3 * 64 * 32 * 32 * 16) / 256, 256, 0, stream>>>(Wfh, Wih, Woh, pw);
  lstm_scan_k<<<NWG, WGSZ, 229376, stream>>>(x, tbl, pw, cb, hT, bar);
  head_k<<<BATCH, 32, 0, stream>>>(hT, Wph, bp, out);
}
